// SimpleTransformerLayer_48756468744619
// MI455X (gfx1250) — matrix operand order repaired and checked by reading, not compiled, not run on hardware
//
#include <hip/hip_runtime.h>
#include <hip/hip_bf16.h>
#include <cstdint>

// ---------------------------------------------------------------------------
// Types for CDNA5 WMMA (wave32): D(16x16,f32) = A(16x32,bf16) * B(32x16,bf16) + C
// ---------------------------------------------------------------------------
typedef __attribute__((ext_vector_type(16))) __bf16 v16bf;
typedef __attribute__((ext_vector_type(8)))  __bf16 v8bf;
typedef __attribute__((ext_vector_type(4)))  __bf16 v4bf;
typedef __attribute__((ext_vector_type(8)))  float  v8f;
typedef __attribute__((ext_vector_type(4)))  unsigned int v4u;
typedef __attribute__((ext_vector_type(8)))  int v8i;
typedef __attribute__((ext_vector_type(4)))  int v4i;

// Model dims (fixed by the reference)
#define HDIM 2048
#define IDIM 8192
#define NHEAD 16
#define HEADD 128
#define SEQ  2048
#define BATCH 2
#define MROWS (BATCH*SEQ)   // 4096

// A-operand fragment: per-lane 16 contiguous bf16 at p[0..15]
__device__ __forceinline__ v16bf frag_gap16(const __bf16* p) {
  v8bf lo = *(const v8bf*)p;
  v8bf hi = *(const v8bf*)(p + 8);
  return __builtin_shufflevector(lo, hi, 0,1,2,3,4,5,6,7,8,9,10,11,12,13,14,15);
}
// B-operand fragment: per-lane 16 contiguous bf16 at p[0..15]
__device__ __forceinline__ v16bf frag_contig(const __bf16* p) {
  v8bf lo = *(const v8bf*)p;
  v8bf hi = *(const v8bf*)(p + 8);
  return __builtin_shufflevector(lo, hi, 0,1,2,3,4,5,6,7,8,9,10,11,12,13,14,15);
}

__device__ __forceinline__ v8f wmma_bf16(v16bf a, v16bf b, v8f c) {
  return __builtin_amdgcn_wmma_f32_16x16x32_bf16(false, a, false, b, (short)0, c, false, false);
}

// ---------------------------------------------------------------------------
// Tensor Data Mover: 2D tile global -> LDS with per-row padding.
//   data_size = 2 bytes (bf16); dims/strides in elements; addresses in bytes.
//   pad_interval_code: pad after 2<<code DWORDs; pad_amount_code: (code+1) DWORDs.
//   (6-arg builtin form on this toolchain: g0, g1, g2, g3, g4-ext, cpol)
// ---------------------------------------------------------------------------
__device__ __forceinline__ void tdm_load_2d(const void* gptr, uint32_t lds_byte,
                                            int d0_stride, int tile_d0, int tile_d1,
                                            int pad_interval_code, int pad_amount_code) {
  const uint64_t ga = (uint64_t)(uintptr_t)gptr;
  v4u g0;
  g0.x = 1u;                                            // count=1 (valid descriptor)
  g0.y = lds_byte;                                      // lds_addr
  g0.z = (uint32_t)(ga & 0xffffffffu);                  // global_addr[31:0]
  g0.w = (uint32_t)((ga >> 32) & 0x01ffffffu) | (2u << 30);  // addr[56:32] | type=2
  v8i g1;
  g1[0] = (int)((1u << 16)                              // data_size = 2B
              | (1u << 20)                              // pad_enable
              | ((uint32_t)pad_interval_code << 22)
              | ((uint32_t)pad_amount_code << 25));
  g1[1] = (int)(((uint32_t)tile_d0 & 0xffffu) << 16);   // tensor_dim0[15:0]
  g1[2] = (int)((((uint32_t)tile_d0 >> 16) & 0xffffu)
              | (((uint32_t)tile_d1 & 0xffffu) << 16)); // tensor_dim0 hi | tensor_dim1 lo
  g1[3] = (int)((((uint32_t)tile_d1 >> 16) & 0xffffu)
              | (((uint32_t)tile_d0 & 0xffffu) << 16)); // tensor_dim1 hi | tile_dim0
  g1[4] = tile_d1 & 0xffff;                             // tile_dim1 (tile_dim2 = 0)
  g1[5] = d0_stride;                                    // tensor_dim0_stride[31:0]
  g1[6] = 0;
  g1[7] = 0;
  const v4i z4 = (v4i){0, 0, 0, 0};
  const v8i z8 = (v8i){0, 0, 0, 0, 0, 0, 0, 0};
  __builtin_amdgcn_tensor_load_to_lds(g0, g1, z4, z4, z8, 0);
}

__device__ __forceinline__ uint32_t lds_off(const void* p) {
  return (uint32_t)(uintptr_t)p;
}

// ---------------------------------------------------------------------------
// fp32 -> bf16 conversion (weights / one-time)
// ---------------------------------------------------------------------------
__launch_bounds__(256)
__global__ void conv_bf16_kernel(const float* __restrict__ x, __bf16* __restrict__ y, long n4) {
  for (long i = (long)blockIdx.x * 256 + threadIdx.x; i < n4; i += (long)gridDim.x * 256) {
    const float4 v = ((const float4*)x)[i];
    v4bf t; t[0] = (__bf16)v.x; t[1] = (__bf16)v.y; t[2] = (__bf16)v.z; t[3] = (__bf16)v.w;
    ((v4bf*)y)[i] = t;
  }
}

// ---------------------------------------------------------------------------
// LayerNorm: fp32 in -> bf16 out, one block per row
// ---------------------------------------------------------------------------
__launch_bounds__(256)
__global__ void ln_kernel(const float* __restrict__ x, const float* __restrict__ w,
                          const float* __restrict__ b, __bf16* __restrict__ y) {
  __shared__ float rs[256];
  __shared__ float rq[256];
  const int tid = threadIdx.x;
  const size_t row = blockIdx.x;
  const float* xr = x + row * HDIM;
  float s = 0.f, q2 = 0.f;
  for (int i = tid; i < HDIM; i += 256) { float v = xr[i]; s += v; q2 += v * v; }
  rs[tid] = s; rq[tid] = q2;
  __syncthreads();
  for (int off = 128; off > 0; off >>= 1) {
    if (tid < off) { rs[tid] += rs[tid + off]; rq[tid] += rq[tid + off]; }
    __syncthreads();
  }
  const float mean = rs[0] * (1.0f / HDIM);
  const float var  = rq[0] * (1.0f / HDIM) - mean * mean;
  const float inv  = rsqrtf(var + 1e-5f);
  __bf16* yr = y + row * HDIM;
  for (int i = tid; i < HDIM; i += 256)
    yr[i] = (__bf16)((xr[i] - mean) * inv * w[i] + b[i]);
}

// ---------------------------------------------------------------------------
// GEMM: C[M,N] = A[M,K](bf16) * W[N,K]^T(bf16) (+bias)(+resid fp32), OutT = float|bf16
// 128x128 tile, BK=32, TDM double-buffered LDS staging.
// ---------------------------------------------------------------------------
#define LDT 40   // padded LDS row stride (bf16): 32 data + 8 pad (16B pad / 64B row)

template <typename OutT>
__launch_bounds__(256)
__global__ void gemm_kernel(const __bf16* __restrict__ A, const __bf16* __restrict__ W,
                            const float* __restrict__ bias, const float* __restrict__ resid,
                            OutT* __restrict__ C, int M, int N, int K) {
  __shared__ __align__(16) __bf16 As[2][128 * LDT];
  __shared__ __align__(16) __bf16 Bs[2][128 * LDT];
  const int tid  = threadIdx.x;
  const int wid  = tid >> 5;
  const int l    = tid & 31;
  const int half = l >> 4;
  const int ln   = l & 15;
  const int bm = blockIdx.y, bn = blockIdx.x;
  const int wm = wid & 3;        // 4 waves on M  -> 32 rows each
  const int wn = wid >> 2;       // 2 waves on N  -> 64 cols each

  v8f acc[2][4];
  #pragma unroll
  for (int mi = 0; mi < 2; mi++)
    #pragma unroll
    for (int nj = 0; nj < 4; nj++) acc[mi][nj] = (v8f){0,0,0,0,0,0,0,0};

  const __bf16* Abase = A + (size_t)(bm * 128) * K;
  const __bf16* Wbase = W + (size_t)(bn * 128) * K;
  const int nsteps = K / 32;

  if (wid == 0) {
    tdm_load_2d(Abase, lds_off(&As[0][0]), K, 32, 128, 3, 3);
    tdm_load_2d(Wbase, lds_off(&Bs[0][0]), K, 32, 128, 3, 3);
  }

  for (int i = 0; i < nsteps; i++) {
    const int buf = i & 1;
    if (wid == 0) {
      if (i + 1 < nsteps) {
        tdm_load_2d(Abase + (i + 1) * 32, lds_off(&As[buf ^ 1][0]), K, 32, 128, 3, 3);
        tdm_load_2d(Wbase + (i + 1) * 32, lds_off(&Bs[buf ^ 1][0]), K, 32, 128, 3, 3);
        __builtin_amdgcn_s_wait_tensorcnt(2);
      } else {
        __builtin_amdgcn_s_wait_tensorcnt(0);
      }
    }
    __syncthreads();

    v16bf bfr[4];
    #pragma unroll
    for (int nj = 0; nj < 4; nj++)
      bfr[nj] = frag_contig(&Bs[buf][(wn * 64 + nj * 16 + ln) * LDT + half * 16]);
    #pragma unroll
    for (int mi = 0; mi < 2; mi++) {
      v16bf afr = frag_gap16(&As[buf][(wm * 32 + mi * 16 + ln) * LDT + half * 16]);
      #pragma unroll
      for (int nj = 0; nj < 4; nj++)
        acc[mi][nj] = wmma_bf16(afr, bfr[nj], acc[mi][nj]);
    }
    __syncthreads();
  }

  // epilogue (C layout: lanes 0-15 -> rows v, lanes 16-31 -> rows v+8; col = ln)
  #pragma unroll
  for (int mi = 0; mi < 2; mi++)
    #pragma unroll
    for (int nj = 0; nj < 4; nj++)
      #pragma unroll
      for (int v = 0; v < 8; v++) {
        const int row = bm * 128 + wm * 32 + mi * 16 + v + half * 8;
        const int col = bn * 128 + wn * 64 + nj * 16 + ln;
        float x = acc[mi][nj][v];
        if (bias)  x += bias[col];
        if (resid) x += resid[(size_t)row * N + col];
        C[(size_t)row * N + col] = (OutT)x;
      }
}

// ---------------------------------------------------------------------------
// Fused gate+up MLP GEMM: Hmid(bf16) = silu(X*Wg^T) * (X*Wu^T), TDM double-buffered
// ---------------------------------------------------------------------------
__launch_bounds__(256)
__global__ void gateup_kernel(const __bf16* __restrict__ X, const __bf16* __restrict__ Wg,
                              const __bf16* __restrict__ Wu, __bf16* __restrict__ Hmid,
                              int M, int N, int K) {
  __shared__ __align__(16) __bf16 As[2][128 * LDT];
  __shared__ __align__(16) __bf16 Gs[2][128 * LDT];
  __shared__ __align__(16) __bf16 Us[2][128 * LDT];
  const int tid  = threadIdx.x;
  const int wid  = tid >> 5;
  const int l    = tid & 31;
  const int half = l >> 4;
  const int ln   = l & 15;
  const int bm = blockIdx.y, bn = blockIdx.x;
  const int wm = wid & 3, wn = wid >> 2;

  v8f ag[2][4], au[2][4];
  #pragma unroll
  for (int mi = 0; mi < 2; mi++)
    #pragma unroll
    for (int nj = 0; nj < 4; nj++) {
      ag[mi][nj] = (v8f){0,0,0,0,0,0,0,0};
      au[mi][nj] = (v8f){0,0,0,0,0,0,0,0};
    }

  const __bf16* Xbase = X  + (size_t)(bm * 128) * K;
  const __bf16* Gbase = Wg + (size_t)(bn * 128) * K;
  const __bf16* Ubase = Wu + (size_t)(bn * 128) * K;
  const int nsteps = K / 32;

  if (wid == 0) {
    tdm_load_2d(Xbase, lds_off(&As[0][0]), K, 32, 128, 3, 3);
    tdm_load_2d(Gbase, lds_off(&Gs[0][0]), K, 32, 128, 3, 3);
    tdm_load_2d(Ubase, lds_off(&Us[0][0]), K, 32, 128, 3, 3);
  }

  for (int i = 0; i < nsteps; i++) {
    const int buf = i & 1;
    if (wid == 0) {
      if (i + 1 < nsteps) {
        tdm_load_2d(Xbase + (i + 1) * 32, lds_off(&As[buf ^ 1][0]), K, 32, 128, 3, 3);
        tdm_load_2d(Gbase + (i + 1) * 32, lds_off(&Gs[buf ^ 1][0]), K, 32, 128, 3, 3);
        tdm_load_2d(Ubase + (i + 1) * 32, lds_off(&Us[buf ^ 1][0]), K, 32, 128, 3, 3);
        __builtin_amdgcn_s_wait_tensorcnt(3);
      } else {
        __builtin_amdgcn_s_wait_tensorcnt(0);
      }
    }
    __syncthreads();

    #pragma unroll
    for (int mi = 0; mi < 2; mi++) {
      v16bf afr = frag_gap16(&As[buf][(wm * 32 + mi * 16 + ln) * LDT + half * 16]);
      #pragma unroll
      for (int nj = 0; nj < 4; nj++) {
        v16bf gfr = frag_contig(&Gs[buf][(wn * 64 + nj * 16 + ln) * LDT + half * 16]);
        v16bf ufr = frag_contig(&Us[buf][(wn * 64 + nj * 16 + ln) * LDT + half * 16]);
        ag[mi][nj] = wmma_bf16(afr, gfr, ag[mi][nj]);
        au[mi][nj] = wmma_bf16(afr, ufr, au[mi][nj]);
      }
    }
    __syncthreads();
  }

  #pragma unroll
  for (int mi = 0; mi < 2; mi++)
    #pragma unroll
    for (int nj = 0; nj < 4; nj++)
      #pragma unroll
      for (int v = 0; v < 8; v++) {
        const int row = bm * 128 + wm * 32 + mi * 16 + v + half * 8;
        const int col = bn * 128 + wn * 64 + nj * 16 + ln;
        const float g = ag[mi][nj][v];
        const float u = au[mi][nj][v];
        Hmid[(size_t)row * N + col] = (__bf16)(g / (1.0f + __expf(-g)) * u);
      }
}

// ---------------------------------------------------------------------------
// Flash attention (bf16 q/k/v in, bf16 out): one block per (batch, head, 128 q rows)
// K tile via TDM (overlapped with manual transposed V copy), online softmax.
// ---------------------------------------------------------------------------
#define KBLK 64
#define KLD 136   // Klds row stride (bf16): 128 data + 8 pad (TDM pads 16B / 256B row)
#define VLD 72    // Vlds row stride:        [hd][kpos] transposed
#define PLD 72    // Plds row stride:        per-wave [16][KBLK]

__launch_bounds__(256)
__global__ void flash_kernel(const __bf16* __restrict__ Qg, const __bf16* __restrict__ Kg,
                             const __bf16* __restrict__ Vg, __bf16* __restrict__ Og) {
  __shared__ __align__(16) __bf16 Klds[KBLK * KLD];
  __shared__ __align__(16) __bf16 Vlds[HEADD * VLD];
  __shared__ __align__(16) __bf16 Plds[8 * 16 * PLD];

  const int tid  = threadIdx.x;
  const int wid  = tid >> 5;
  const int l    = tid & 31;
  const int half = l >> 4;
  const int ln   = l & 15;

  const int nqb = SEQ / 128;                  // 16
  const int qb  = blockIdx.x % nqb;
  const int h   = (blockIdx.x / nqb) % NHEAD;
  const int bb  = blockIdx.x / (nqb * NHEAD);
  const size_t base = (size_t)bb * SEQ * HDIM + (size_t)h * HEADD;
  const float scale = 0.08838834764831845f;   // 1/sqrt(128)

  // Q fragments in registers (A-operand layout)
  v16bf qf[4];
  {
    const int qrow = qb * 128 + wid * 16 + ln;
    const __bf16* qp = Qg + base + (size_t)qrow * HDIM;
    #pragma unroll
    for (int kk = 0; kk < 4; kk++) {
      const int K0 = kk * 32 + half * 16;
      qf[kk] = frag_gap16(qp + K0);
    }
  }

  float m_i[8], l_i[8];
  v8f   oacc[8];
  #pragma unroll
  for (int v = 0; v < 8; v++) { m_i[v] = -1e30f; l_i[v] = 0.f; }
  #pragma unroll
  for (int no = 0; no < 8; no++) oacc[no] = (v8f){0,0,0,0,0,0,0,0};

  __bf16* Pw = &Plds[wid * 16 * PLD];

  for (int kb = 0; kb < SEQ; kb += KBLK) {
    // K tile (64 x 128, row-major, padded) via TDM
    if (wid == 0)
      tdm_load_2d(Kg + base + (size_t)kb * HDIM, lds_off(&Klds[0]), HDIM, 128, KBLK, 5, 3);

    // V tile transposed into LDS (batched loads, then scattered stores)
    v8bf vr[4];
    #pragma unroll
    for (int i = 0; i < 4; i++) {
      const int u  = tid + 256 * i;     // 0..1023
      const int r  = u >> 4;            // 0..63 (kpos)
      const int c8 = u & 15;            // 8-bf16 chunk along hd
      vr[i] = *(const v8bf*)(Vg + base + (size_t)(kb + r) * HDIM + c8 * 8);
    }
    #pragma unroll
    for (int i = 0; i < 4; i++) {
      const int u  = tid + 256 * i;
      const int r  = u >> 4;
      const int c8 = u & 15;
      #pragma unroll
      for (int j = 0; j < 8; j++)
        Vlds[(c8 * 8 + j) * VLD + r] = vr[i][j];
    }

    if (wid == 0) __builtin_amdgcn_s_wait_tensorcnt(0);
    __syncthreads();

    // S strip (16 x 64) = Q * K^T
    v8f sa[4];
    #pragma unroll
    for (int nj = 0; nj < 4; nj++) {
      v8f c = (v8f){0,0,0,0,0,0,0,0};
      #pragma unroll
      for (int kk = 0; kk < 4; kk++) {
        v16bf bfr = frag_contig(&Klds[(nj * 16 + ln) * KLD + kk * 32 + half * 16]);
        c = wmma_bf16(qf[kk], bfr, c);
      }
      sa[nj] = c;
    }

    // online softmax (rows 0-7 in lanes 0-15, rows 8-15 in lanes 16-31)
    #pragma unroll
    for (int v = 0; v < 8; v++) {
      float mx = -1e30f;
      #pragma unroll
      for (int nj = 0; nj < 4; nj++) { sa[nj][v] *= scale; mx = fmaxf(mx, sa[nj][v]); }
      for (int off = 1; off < 16; off <<= 1) mx = fmaxf(mx, __shfl_xor(mx, off, 16));
      const float mnew = fmaxf(m_i[v], mx);
      const float corr = __expf(m_i[v] - mnew);
      float rs = 0.f;
      #pragma unroll
      for (int nj = 0; nj < 4; nj++) {
        const float p = __expf(sa[nj][v] - mnew);
        sa[nj][v] = p;
        rs += p;
      }
      for (int off = 1; off < 16; off <<= 1) rs += __shfl_xor(rs, off, 16);
      l_i[v] = l_i[v] * corr + rs;
      m_i[v] = mnew;
      #pragma unroll
      for (int no = 0; no < 8; no++) oacc[no][v] *= corr;
    }

    // re-layout P: C-layout -> LDS -> A-operand fragments
    #pragma unroll
    for (int nj = 0; nj < 4; nj++)
      #pragma unroll
      for (int v = 0; v < 8; v++)
        Pw[(v + half * 8) * PLD + nj * 16 + ln] = (__bf16)sa[nj][v];

    // O (16 x 128) += P (16 x 64) * V (64 x 128)
    #pragma unroll
    for (int kk = 0; kk < 2; kk++) {
      v16bf pf = frag_gap16(&Pw[ln * PLD + kk * 32 + half * 16]);
      #pragma unroll
      for (int no = 0; no < 8; no++) {
        v16bf vf = frag_contig(&Vlds[(no * 16 + ln) * VLD + kk * 32 + half * 16]);
        oacc[no] = wmma_bf16(pf, vf, oacc[no]);
      }
    }
    __syncthreads();
  }

  // epilogue: normalize, store bf16 at [s, h*HD + d]
  #pragma unroll
  for (int no = 0; no < 8; no++)
    #pragma unroll
    for (int v = 0; v < 8; v++) {
      const int row = qb * 128 + wid * 16 + v + half * 8;
      const int col = no * 16 + ln;
      Og[base + (size_t)row * HDIM + col] = (__bf16)(oacc[no][v] / l_i[v]);
    }
}

// ---------------------------------------------------------------------------
// Orchestration
// ---------------------------------------------------------------------------
extern "C" void kernel_launch(void* const* d_in, const int* in_sizes, int n_in,
                              void* d_out, int out_size, void* d_ws, size_t ws_size,
                              hipStream_t stream) {
  const float* hs   = (const float*)d_in[0];
  const float* Wq   = (const float*)d_in[1];
  const float* bq   = (const float*)d_in[2];
  const float* Wk   = (const float*)d_in[3];
  const float* bk   = (const float*)d_in[4];
  const float* Wv   = (const float*)d_in[5];
  const float* bv   = (const float*)d_in[6];
  const float* Wo   = (const float*)d_in[7];
  const float* bo   = (const float*)d_in[8];
  const float* Wg   = (const float*)d_in[9];
  const float* Wu   = (const float*)d_in[10];
  const float* Wd   = (const float*)d_in[11];
  const float* ln1w = (const float*)d_in[12];
  const float* ln1b = (const float*)d_in[13];
  const float* ln2w = (const float*)d_in[14];
  const float* ln2b = (const float*)d_in[15];
  float* out = (float*)d_out;

  const size_t HH = (size_t)HDIM * HDIM;     // 4.19M
  const size_t HI = (size_t)HDIM * IDIM;     // 16.8M
  const size_t MH = (size_t)MROWS * HDIM;    // 8.39M
  const size_t MI = (size_t)MROWS * IDIM;    // 33.6M

  char* p = (char*)d_ws;
  auto carve = [&](size_t bytes) { char* r = p; p += (bytes + 255) & ~(size_t)255; return (void*)r; };
  __bf16* WqB   = (__bf16*)carve(HH * 2);
  __bf16* WkB   = (__bf16*)carve(HH * 2);
  __bf16* WvB   = (__bf16*)carve(HH * 2);
  __bf16* WoB   = (__bf16*)carve(HH * 2);
  __bf16* WgB   = (__bf16*)carve(HI * 2);
  __bf16* WuB   = (__bf16*)carve(HI * 2);
  __bf16* WdB   = (__bf16*)carve(HI * 2);
  __bf16* xln   = (__bf16*)carve(MH * 2);    // LN1 then LN2 output
  __bf16* qB    = (__bf16*)carve(MH * 2);
  __bf16* kB    = (__bf16*)carve(MH * 2);
  __bf16* vB    = (__bf16*)carve(MH * 2);
  __bf16* attnB = (__bf16*)carve(MH * 2);
  float*  h1    = (float*) carve(MH * 4);
  __bf16* hmidB = (__bf16*)carve(MI * 2);

  const dim3 blk(256);
  const dim3 g_hh(HDIM / 128, MROWS / 128);  // (16, 32)
  const dim3 g_gu(IDIM / 128, MROWS / 128);  // (64, 32)

  // one-time weight conversion fp32 -> bf16
  conv_bf16_kernel<<<2048, blk, 0, stream>>>(Wq, WqB, (long)(HH / 4));
  conv_bf16_kernel<<<2048, blk, 0, stream>>>(Wk, WkB, (long)(HH / 4));
  conv_bf16_kernel<<<2048, blk, 0, stream>>>(Wv, WvB, (long)(HH / 4));
  conv_bf16_kernel<<<2048, blk, 0, stream>>>(Wo, WoB, (long)(HH / 4));
  conv_bf16_kernel<<<4096, blk, 0, stream>>>(Wg, WgB, (long)(HI / 4));
  conv_bf16_kernel<<<4096, blk, 0, stream>>>(Wu, WuB, (long)(HI / 4));
  conv_bf16_kernel<<<4096, blk, 0, stream>>>(Wd, WdB, (long)(HI / 4));

  // LN1
  ln_kernel<<<MROWS, blk, 0, stream>>>(hs, ln1w, ln1b, xln);
  // Q, K, V projections (bf16 out)
  gemm_kernel<__bf16><<<g_hh, blk, 0, stream>>>(xln, WqB, bq, nullptr, qB, MROWS, HDIM, HDIM);
  gemm_kernel<__bf16><<<g_hh, blk, 0, stream>>>(xln, WkB, bk, nullptr, kB, MROWS, HDIM, HDIM);
  gemm_kernel<__bf16><<<g_hh, blk, 0, stream>>>(xln, WvB, bv, nullptr, vB, MROWS, HDIM, HDIM);
  // attention
  flash_kernel<<<BATCH * NHEAD * (SEQ / 128), blk, 0, stream>>>(qB, kB, vB, attnB);
  // output projection + residual (fp32 out)
  gemm_kernel<float><<<g_hh, blk, 0, stream>>>(attnB, WoB, bo, hs, h1, MROWS, HDIM, HDIM);
  // LN2
  ln_kernel<<<MROWS, blk, 0, stream>>>(h1, ln2w, ln2b, xln);
  // fused gate+up with SiLU (bf16 out)
  gateup_kernel<<<g_gu, blk, 0, stream>>>(xln, WgB, WuB, hmidB, MROWS, IDIM, HDIM);
  // down projection + residual -> fp32 output
  gemm_kernel<float><<<g_hh, blk, 0, stream>>>(hmidB, WdB, nullptr, h1, out, MROWS, HDIM, IDIM);
}
